// Transformer_62929860821237
// MI455X (gfx1250) — matrix operand order repaired and checked by reading, not compiled, not run on hardware
//
#include <hip/hip_runtime.h>
#include <hip/hip_bf16.h>

// ---------------------------------------------------------------------------
// CDNA5 (gfx1250) transformer forward: all GEMMs + attention via
// v_wmma_f32_16x16x32_f16 (wave32), async global->LDS staging (ASYNCcnt),
// double-buffered LDS, f32 accumulate, f32 residual stream.
// ---------------------------------------------------------------------------

typedef __attribute__((ext_vector_type(16))) _Float16 v16h;
typedef __attribute__((ext_vector_type(8)))  float    v8f;
typedef __attribute__((ext_vector_type(4)))  int      v4i;

union Frag { uint4 u[2]; v16h h; };

#define LDS_STRIDE 40  // halves; 80B rows keep 16B alignment

#define AS1 __attribute__((address_space(1)))
#define AS3 __attribute__((address_space(3)))

// 16B global -> LDS copy. Prefers the CDNA5 async-DMA path (ASYNCcnt).
__device__ __forceinline__ void async_copy16(const _Float16* g, _Float16* l) {
#if defined(__gfx1250__) && __has_builtin(__builtin_amdgcn_global_load_async_to_lds_b128)
  __builtin_amdgcn_global_load_async_to_lds_b128(
      (AS1 v4i*)(g), (AS3 v4i*)(l), 0, 0);
#else
  *(uint4*)l = *(const uint4*)g;  // synchronous fallback
#endif
}

template <int N>
__device__ __forceinline__ void async_wait() {
#if defined(__gfx1250__) && __has_builtin(__builtin_amdgcn_s_wait_asynccnt)
  __builtin_amdgcn_s_wait_asynccnt(N);
#endif
}

// ---------------------------------------------------------------------------
// f32 [K,N] weight -> f16 [N,K] transposed copy (so B-tiles are K-contiguous)
// ---------------------------------------------------------------------------
__global__ void wt_kernel(const float* __restrict__ W, _Float16* __restrict__ Wt,
                          int K, int N) {
  int idx = blockIdx.x * blockDim.x + threadIdx.x;
  if (idx >= K * N) return;
  int k = idx / N, n = idx - k * N;
  Wt[(size_t)n * K + k] = (_Float16)W[idx];
}

// ---------------------------------------------------------------------------
// Embedding gather: out[b*S+s] = tokEmb[tok[b*S+s]] + posEmb[s]
// ---------------------------------------------------------------------------
__global__ void embed_kernel(const int* __restrict__ tok,
                             const float* __restrict__ tokEmb,
                             const float* __restrict__ posEmb,
                             float* __restrict__ out, int S) {
  int row = blockIdx.x;                 // b*S + s
  int s   = row & (S - 1);
  int id  = tok[row];
  const float* te = tokEmb + (size_t)id * 512;
  const float* pe = posEmb + (size_t)s  * 512;
  float* o = out + (size_t)row * 512;
  for (int c = threadIdx.x; c < 512; c += blockDim.x) o[c] = te[c] + pe[c];
}

// ---------------------------------------------------------------------------
// LayerNorm over 512 columns: 1 wave per row, dual f32 / packed-f16 output
// ---------------------------------------------------------------------------
__global__ __launch_bounds__(256) void ln_kernel(
    const float* __restrict__ x, const float* __restrict__ g,
    const float* __restrict__ bb, float* __restrict__ outF,
    _Float16* __restrict__ outH, int nrows) {
  int wave = threadIdx.x >> 5, lane = threadIdx.x & 31;
  int row = blockIdx.x * 8 + wave;
  if (row >= nrows) return;
  const float* xr = x + (size_t)row * 512;
  float4 v[4];
  float s = 0.f, s2 = 0.f;
#pragma unroll
  for (int i = 0; i < 4; ++i) {
    v[i] = *(const float4*)(xr + lane * 16 + i * 4);
    s  += v[i].x + v[i].y + v[i].z + v[i].w;
    s2 += v[i].x * v[i].x + v[i].y * v[i].y + v[i].z * v[i].z + v[i].w * v[i].w;
  }
#pragma unroll
  for (int off = 16; off > 0; off >>= 1) {
    s  += __shfl_xor(s, off, 32);
    s2 += __shfl_xor(s2, off, 32);
  }
  float mean = s * (1.f / 512.f);
  float var  = s2 * (1.f / 512.f) - mean * mean;
  float rs   = rsqrtf(var + 1e-5f);
  union { _Float16 h[16]; uint4 u[2]; } hv;
#pragma unroll
  for (int i = 0; i < 4; ++i) {
    int c = lane * 16 + i * 4;
    float y0 = (v[i].x - mean) * rs * g[c + 0] + bb[c + 0];
    float y1 = (v[i].y - mean) * rs * g[c + 1] + bb[c + 1];
    float y2 = (v[i].z - mean) * rs * g[c + 2] + bb[c + 2];
    float y3 = (v[i].w - mean) * rs * g[c + 3] + bb[c + 3];
    if (outF) *(float4*)(outF + (size_t)row * 512 + c) = make_float4(y0, y1, y2, y3);
    hv.h[i * 4 + 0] = (_Float16)y0; hv.h[i * 4 + 1] = (_Float16)y1;
    hv.h[i * 4 + 2] = (_Float16)y2; hv.h[i * 4 + 3] = (_Float16)y3;
  }
  if (outH) {
    *(uint4*)(outH + (size_t)row * 512 + lane * 16)     = hv.u[0];
    *(uint4*)(outH + (size_t)row * 512 + lane * 16 + 8) = hv.u[1];
  }
}

// ---------------------------------------------------------------------------
// GEMM: out[M,N] = A[M,K](f16) @ Wt[N,K]^T(f16) + bias, optional residual /
// ReLU, dual f32/f16 outputs. Block tile 128x128x32, 8 waves, wave tile 32x64.
// Double-buffered LDS, async global->LDS staging.
// ---------------------------------------------------------------------------
__global__ __launch_bounds__(256) void gemm_kernel(
    const _Float16* __restrict__ A, const _Float16* __restrict__ Wt,
    const float* __restrict__ bias, const float* __restrict__ res,
    float* __restrict__ outF, _Float16* __restrict__ outH,
    int M, int N, int K, int relu) {
  __shared__ __align__(16) _Float16 As[2][128 * LDS_STRIDE];
  __shared__ __align__(16) _Float16 Bs[2][128 * LDS_STRIDE];
  const int tid  = threadIdx.x;
  const int lane = tid & 31;
  const int wave = tid >> 5;
  const int laneLo = lane & 15, laneHi = lane >> 4;
  const int wm = wave & 3, wn = wave >> 2;           // 4 x 2 wave grid
  const int blockM = blockIdx.x * 128, blockN = blockIdx.y * 128;
  const int waveM = wm * 32, waveN = wn * 64;

  // fixed per-thread staging coordinates (2 x 16B segments per matrix)
  const int r0 = tid >> 2,          c0 = (tid & 3) * 8;
  const int r1 = (256 + tid) >> 2,  c1 = ((256 + tid) & 3) * 8;

  v8f acc[2][4] = {};
  const int nT = K >> 5;

  auto stage = [&](int buf, int k0) {
    async_copy16(A  + (size_t)(blockM + r0) * K + k0 + c0, &As[buf][r0 * LDS_STRIDE + c0]);
    async_copy16(A  + (size_t)(blockM + r1) * K + k0 + c1, &As[buf][r1 * LDS_STRIDE + c1]);
    async_copy16(Wt + (size_t)(blockN + r0) * K + k0 + c0, &Bs[buf][r0 * LDS_STRIDE + c0]);
    async_copy16(Wt + (size_t)(blockN + r1) * K + k0 + c1, &Bs[buf][r1 * LDS_STRIDE + c1]);
  };

  stage(0, 0);
  for (int t = 0; t < nT; ++t) {
    const int buf = t & 1;
    if (t + 1 < nT) {
      stage(buf ^ 1, (t + 1) * 32);       // prefetch next slab via async DMA
      __builtin_prefetch(A  + (size_t)(blockM + r0) * K + (t + 1) * 32 + c0, 0, 1);
      async_wait<4>();                    // drain only slab t (in-order)
    } else {
      async_wait<0>();
    }
    __syncthreads();

    Frag af[2], bf[4];
#pragma unroll
    for (int mi = 0; mi < 2; ++mi) {  // A frag: lane=row, halves = K octets
      int rr = waveM + mi * 16 + laneLo;
      af[mi].u[0] = *(const uint4*)(&As[buf][rr * LDS_STRIDE + laneHi * 8]);
      af[mi].u[1] = *(const uint4*)(&As[buf][rr * LDS_STRIDE + 16 + laneHi * 8]);
    }
#pragma unroll
    for (int ni = 0; ni < 4; ++ni) {  // B frag: lane=col, halves = K octets
      int rr = waveN + ni * 16 + laneLo;
      bf[ni].u[0] = *(const uint4*)(&Bs[buf][rr * LDS_STRIDE + laneHi * 8]);
      bf[ni].u[1] = *(const uint4*)(&Bs[buf][rr * LDS_STRIDE + 16 + laneHi * 8]);
    }
#pragma unroll
    for (int mi = 0; mi < 2; ++mi)
#pragma unroll
      for (int ni = 0; ni < 4; ++ni)
        acc[mi][ni] = __builtin_amdgcn_wmma_f32_16x16x32_f16(
            false, af[mi].h, false, bf[ni].h, (short)0, acc[mi][ni], false, false);
    __syncthreads();
  }

  // epilogue: C layout = VGPR r -> M=r (+8 for hi lanes), lane -> N
#pragma unroll
  for (int ni = 0; ni < 4; ++ni) {
    int col = blockN + waveN + ni * 16 + laneLo;
    float bc = bias ? bias[col] : 0.f;
#pragma unroll
    for (int mi = 0; mi < 2; ++mi) {
#pragma unroll
      for (int r = 0; r < 8; ++r) {
        int row = blockM + waveM + mi * 16 + r + laneHi * 8;
        float v = acc[mi][ni][r] + bc;
        if (res)  v += res[(size_t)row * N + col];
        if (relu) v = fmaxf(v, 0.f);
        if (outF) outF[(size_t)row * N + col] = v;
        if (outH) outH[(size_t)row * N + col] = (_Float16)v;
      }
    }
  }
}

// ---------------------------------------------------------------------------
// Flash attention: 1 wave per (b, h, 16-query tile). dh=64, D=512.
// Per 32-key chunk: V async-staged to LDS (overlaps QK^T), QK^T = 4 WMMAs,
// online softmax, PV = 4 WMMAs.
// ---------------------------------------------------------------------------
__global__ __launch_bounds__(32) void attn_kernel(
    const _Float16* __restrict__ Qh, const _Float16* __restrict__ Kh,
    const _Float16* __restrict__ Vh, _Float16* __restrict__ outH,
    int Lq, int Lk, int causal) {
  __shared__ __align__(16) _Float16 Pl[16 * LDS_STRIDE];  // P tile  16x32
  __shared__ __align__(16) _Float16 Vl[32 * 64];          // V chunk 32x64 (linear)
  const int lane = threadIdx.x & 31;
  const int laneLo = lane & 15, laneHi = lane >> 4;
  const int q0 = blockIdx.x * 16;
  const int h  = blockIdx.y;
  const int b  = blockIdx.z;
  const _Float16* qbase = Qh + ((size_t)b * Lq) * 512 + h * 64;
  const _Float16* kbase = Kh + ((size_t)b * Lk) * 512 + h * 64;
  const _Float16* vbase = Vh + ((size_t)b * Lk) * 512 + h * 64;

  // Q fragments for d-chunks [0,32) and [32,64): held for the whole sweep
  Frag aq[2];
  {
    const _Float16* qrow = qbase + (size_t)(q0 + laneLo) * 512;
#pragma unroll
    for (int c = 0; c < 2; ++c) {
      aq[c].u[0] = *(const uint4*)(qrow + c * 32 + laneHi * 8);
      aq[c].u[1] = *(const uint4*)(qrow + c * 32 + 16 + laneHi * 8);
    }
  }

  float mrow[8], lrow[8];
#pragma unroll
  for (int r = 0; r < 8; ++r) { mrow[r] = -1e30f; lrow[r] = 0.f; }
  v8f ao[4] = {};  // 16x64 output accumulator

  const int kcMax = causal ? ((q0 + 15) >> 5) : ((Lk - 1) >> 5);
  for (int kc = 0; kc <= kcMax; ++kc) {
    // ---- async-stage V chunk (32 keys x 64 d) linearly; overlaps QK^T ----
#pragma unroll
    for (int i = 0; i < 8; ++i) {
      int seg = i * 32 + lane;            // 256 x 16B segments
      int row = seg >> 3, c8 = (seg & 7) * 8;
      async_copy16(vbase + (size_t)(kc * 32 + row) * 512 + c8, &Vl[row * 64 + c8]);
    }
    // ---- scores S = Q K^T for 32 keys (two 16-key column tiles) ----
    v8f sc[2] = {};
#pragma unroll
    for (int nf = 0; nf < 2; ++nf) {
      int key = kc * 32 + nf * 16 + laneLo;
      const _Float16* krow = kbase + (size_t)key * 512;
#pragma unroll
      for (int c = 0; c < 2; ++c) {   // K^T B-frag: K-contiguous in memory
        Frag bk;
        bk.u[0] = *(const uint4*)(krow + c * 32 + laneHi * 8);
        bk.u[1] = *(const uint4*)(krow + c * 32 + 16 + laneHi * 8);
        sc[nf] = __builtin_amdgcn_wmma_f32_16x16x32_f16(
            false, aq[c].h, false, bk.h, (short)0, sc[nf], false, false);
      }
    }
    // ---- online softmax (rows live across 16-lane groups) ----
    float pf0[8], pf1[8];
#pragma unroll
    for (int r = 0; r < 8; ++r) {
      int rowg = q0 + r + laneHi * 8;
      float s0 = sc[0][r] * 0.125f, s1 = sc[1][r] * 0.125f;  // 1/sqrt(64)
      if (causal) {
        int c0 = kc * 32 + laneLo;
        if (c0 > rowg)      s0 = -1e30f;
        if (c0 + 16 > rowg) s1 = -1e30f;
      }
      float mx = fmaxf(s0, s1);
#pragma unroll
      for (int off = 1; off < 16; off <<= 1) mx = fmaxf(mx, __shfl_xor(mx, off, 16));
      float mnew = fmaxf(mrow[r], mx);
      float corr = __expf(mrow[r] - mnew);
      float p0 = __expf(s0 - mnew), p1 = __expf(s1 - mnew);
      float ps = p0 + p1;
#pragma unroll
      for (int off = 1; off < 16; off <<= 1) ps += __shfl_xor(ps, off, 16);
      lrow[r] = lrow[r] * corr + ps;
      mrow[r] = mnew;
#pragma unroll
      for (int ni = 0; ni < 4; ++ni) ao[ni][r] *= corr;
      pf0[r] = p0; pf1[r] = p1;
    }
    // ---- P (C layout) -> LDS -> A layout ----
#pragma unroll
    for (int r = 0; r < 8; ++r) {
      int rl = r + laneHi * 8;
      Pl[rl * LDS_STRIDE + laneLo]      = (_Float16)pf0[r];
      Pl[rl * LDS_STRIDE + 16 + laneLo] = (_Float16)pf1[r];
    }
    async_wait<0>();     // V chunk resident in LDS
    __syncthreads();
    Frag pa;
    pa.u[0] = *(const uint4*)(&Pl[laneLo * LDS_STRIDE + laneHi * 8]);
    pa.u[1] = *(const uint4*)(&Pl[laneLo * LDS_STRIDE + 16 + laneHi * 8]);
#pragma unroll
    for (int ni = 0; ni < 4; ++ni) {
      int d = ni * 16 + laneLo;
      Frag bv;                           // B frag: lane=col(d), k strided rows
#pragma unroll
      for (int e = 0; e < 16; ++e) bv.h[e] = Vl[(laneHi * 8 + (e & 7) + (e >> 3) * 16) * 64 + d];
      ao[ni] = __builtin_amdgcn_wmma_f32_16x16x32_f16(
          false, pa.h, false, bv.h, (short)0, ao[ni], false, false);
    }
    __syncthreads();
  }
  // ---- normalize + write f16 ----
#pragma unroll
  for (int ni = 0; ni < 4; ++ni) {
    int d = ni * 16 + laneLo;
#pragma unroll
    for (int r = 0; r < 8; ++r) {
      int rowg = q0 + r + laneHi * 8;
      float v = ao[ni][r] / lrow[r];
      outH[((size_t)(b * Lq + rowg)) * 512 + h * 64 + d] = (_Float16)v;
    }
  }
}

// ---------------------------------------------------------------------------
// Host orchestration
// ---------------------------------------------------------------------------
extern "C" void kernel_launch(void* const* d_in, const int* in_sizes, int n_in,
                              void* d_out, int out_size, void* d_ws, size_t ws_size,
                              hipStream_t stream) {
  (void)in_sizes; (void)n_in; (void)out_size; (void)ws_size;
  const int Bb = 8, Ss = 512, Tt = 512;
  const int MS = Bb * Ss;  // 4096 rows, same for encoder & decoder

  // ----- flattened pytree indices -----
  // top level (insertion order): src, tgt, src_mask, tgt_mask, params
  // params (sorted): dec[6], dec_norm, enc[6], enc_norm, src_pos, src_tok,
  //                  tgt_pos, tgt_tok ; leaves sorted (b<g, b<w, k<o<q<v ...)
  auto P = [&](int i) -> const float* { return (const float*)d_in[i]; };
  const int* src = (const int*)d_in[0];
  const int* tgt = (const int*)d_in[1];
  const int DEC0 = 4;
  auto dec_idx = [&](int L, int off) { return DEC0 + L * 26 + off; };
  const int DECN = DEC0 + 6 * 26;   // dec_norm.b / .g
  const int ENC0 = DECN + 2;
  auto enc_idx = [&](int L, int off) { return ENC0 + L * 16 + off; };
  const int ENCN = ENC0 + 6 * 16;   // enc_norm.b / .g
  const int SRC_POS = ENCN + 2, SRC_TOK = ENCN + 3;
  const int TGT_POS = ENCN + 4, TGT_TOK = ENCN + 5;

  // ----- workspace bump allocator -----
  char* wp = (char*)d_ws;
  auto alloc = [&](size_t bytes) -> void* {
    void* r = wp; wp += (bytes + 255) & ~(size_t)255; return r;
  };
  float*     x    = (float*)alloc((size_t)MS * 512 * 4);   // encoder stream
  float*     y    = (float*)alloc((size_t)MS * 512 * 4);   // decoder stream
  _Float16*  hh   = (_Float16*)alloc((size_t)MS * 512 * 2);
  _Float16*  qh   = (_Float16*)alloc((size_t)MS * 512 * 2);
  _Float16*  kh   = (_Float16*)alloc((size_t)MS * 512 * 2);
  _Float16*  vh   = (_Float16*)alloc((size_t)MS * 512 * 2);
  _Float16*  ah   = (_Float16*)alloc((size_t)MS * 512 * 2);
  _Float16*  f1h  = (_Float16*)alloc((size_t)MS * 1024 * 2);
  _Float16*  memh = (_Float16*)alloc((size_t)MS * 512 * 2);
  _Float16*  wt   = (_Float16*)alloc((size_t)1024 * 1024 * 2);

  auto gemm = [&](const _Float16* Ain, int wIdx, int bIdx, const float* resid,
                  float* outF, _Float16* outHh, int N, int K, int relu) {
    wt_kernel<<<(K * N + 255) / 256, 256, 0, stream>>>(P(wIdx), wt, K, N);
    dim3 g(MS / 128, N / 128);
    gemm_kernel<<<g, 256, 0, stream>>>(Ain, wt, P(bIdx), resid, outF, outHh,
                                       MS, N, K, relu);
  };
  auto ln = [&](const float* xin, int bIdx, int gIdx, float* outF,
                _Float16* outHh) {
    ln_kernel<<<MS / 8, 256, 0, stream>>>(xin, P(gIdx), P(bIdx), outF, outHh, MS);
  };
  auto attn = [&](const _Float16* q, const _Float16* k, const _Float16* v,
                  _Float16* o, int causal) {
    attn_kernel<<<dim3(Ss / 16, 8, Bb), 32, 0, stream>>>(q, k, v, o, Ss, Ss, causal);
  };

  // ===================== encoder =====================
  embed_kernel<<<MS, 256, 0, stream>>>(src, P(SRC_TOK), P(SRC_POS), x, Ss);
  for (int L = 0; L < 6; ++L) {
    // enc layer leaves: l1.b=0 l1.w=1 l2.b=2 l2.w=3 n1.b=4 n1.g=5 n2.b=6 n2.g=7
    //                   sa.k.b=8 sa.k.w=9 sa.o.b=10 sa.o.w=11 sa.q.b=12 sa.q.w=13
    //                   sa.v.b=14 sa.v.w=15
    ln(x, enc_idx(L, 4), enc_idx(L, 5), nullptr, hh);
    gemm(hh, enc_idx(L, 13), enc_idx(L, 12), nullptr, nullptr, qh, 512, 512, 0);
    gemm(hh, enc_idx(L, 9),  enc_idx(L, 8),  nullptr, nullptr, kh, 512, 512, 0);
    gemm(hh, enc_idx(L, 15), enc_idx(L, 14), nullptr, nullptr, vh, 512, 512, 0);
    attn(qh, kh, vh, ah, 0);
    gemm(ah, enc_idx(L, 11), enc_idx(L, 10), x, x, nullptr, 512, 512, 0);
    ln(x, enc_idx(L, 6), enc_idx(L, 7), nullptr, hh);
    gemm(hh,  enc_idx(L, 1), enc_idx(L, 0), nullptr, nullptr, f1h, 1024, 512, 1);
    gemm(f1h, enc_idx(L, 3), enc_idx(L, 2), x, x, nullptr, 512, 1024, 0);
  }
  ln(x, ENCN + 0, ENCN + 1, nullptr, memh);  // memory (f16 feed for cross-attn)

  // ===================== decoder =====================
  embed_kernel<<<MS, 256, 0, stream>>>(tgt, P(TGT_TOK), P(TGT_POS), y, Tt);
  for (int L = 0; L < 6; ++L) {
    // dec layer leaves: ca.k.b=0 ca.k.w=1 ca.o.b=2 ca.o.w=3 ca.q.b=4 ca.q.w=5
    //   ca.v.b=6 ca.v.w=7 l1.b=8 l1.w=9 l2.b=10 l2.w=11 n1.b=12 n1.g=13
    //   n2.b=14 n2.g=15 n3.b=16 n3.g=17 sa.k.b=18 sa.k.w=19 sa.o.b=20 sa.o.w=21
    //   sa.q.b=22 sa.q.w=23 sa.v.b=24 sa.v.w=25
    ln(y, dec_idx(L, 12), dec_idx(L, 13), nullptr, hh);
    gemm(hh, dec_idx(L, 23), dec_idx(L, 22), nullptr, nullptr, qh, 512, 512, 0);
    gemm(hh, dec_idx(L, 19), dec_idx(L, 18), nullptr, nullptr, kh, 512, 512, 0);
    gemm(hh, dec_idx(L, 25), dec_idx(L, 24), nullptr, nullptr, vh, 512, 512, 0);
    attn(qh, kh, vh, ah, 1);  // causal self-attention
    gemm(ah, dec_idx(L, 21), dec_idx(L, 20), y, y, nullptr, 512, 512, 0);
    ln(y, dec_idx(L, 14), dec_idx(L, 15), nullptr, hh);
    gemm(hh,   dec_idx(L, 5), dec_idx(L, 4), nullptr, nullptr, qh, 512, 512, 0);
    gemm(memh, dec_idx(L, 1), dec_idx(L, 0), nullptr, nullptr, kh, 512, 512, 0);
    gemm(memh, dec_idx(L, 7), dec_idx(L, 6), nullptr, nullptr, vh, 512, 512, 0);
    attn(qh, kh, vh, ah, 0);  // cross-attention (src_mask is all-ones)
    gemm(ah, dec_idx(L, 3), dec_idx(L, 2), y, y, nullptr, 512, 512, 0);
    ln(y, dec_idx(L, 16), dec_idx(L, 17), nullptr, hh);
    gemm(hh,  dec_idx(L, 9),  dec_idx(L, 8),  nullptr, nullptr, f1h, 1024, 512, 1);
    gemm(f1h, dec_idx(L, 11), dec_idx(L, 10), y, y, nullptr, 512, 1024, 0);
  }
  ln(y, DECN + 0, DECN + 1, (float*)d_out, hh);  // final dec_norm -> f32 out
}
